// PatchAttention_65154653880395
// MI455X (gfx1250) — matrix operand order repaired and checked by reading, not compiled, not run on hardware
//
#include <hip/hip_runtime.h>

// ---------------------------------------------------------------------------
// Problem constants (from reference): B=2, N=2048, D=1024, H=16, CD=3, HD=64
// ---------------------------------------------------------------------------
#define BATCH 2
#define SEQ   2048
#define DMODEL 1024
#define NHEAD 16
#define HDIM  64

typedef __bf16 bf16;
typedef __attribute__((ext_vector_type(4)))  __bf16 v4bf;
typedef __attribute__((ext_vector_type(8)))  __bf16 v8bf;
typedef __attribute__((ext_vector_type(16))) __bf16 v16bf;
typedef __attribute__((ext_vector_type(8)))  float  v8f;
typedef __attribute__((ext_vector_type(4)))  unsigned int u32x4;
typedef __attribute__((ext_vector_type(4)))  int i32x4;
typedef __attribute__((ext_vector_type(8)))  int i32x8;

#if defined(__has_builtin)
#  if __has_builtin(__builtin_amdgcn_tensor_load_to_lds)
#    define HAVE_TDM 1
#  endif
#endif
#if defined(__has_include)
#  if __has_include(<hip/amd_detail/amd_gfx1250_TDM.h>)
#    define TDM_SIX_ARGS 1
#  endif
#endif

__device__ __forceinline__ v16bf bcat(v8bf a, v8bf b) {
    return __builtin_shufflevector(a, b, 0,1,2,3,4,5,6,7,8,9,10,11,12,13,14,15);
}
__device__ __forceinline__ v8bf ld8(const bf16* p) { return *(const v8bf*)p; }

__device__ __forceinline__ v8f wmma_bf16(v16bf a, v16bf b, v8f c) {
    return __builtin_amdgcn_wmma_f32_16x16x32_bf16(false, a, false, b,
                                                   (short)0, c, false, false);
}

// Reductions across a 16-lane half-wave (wave32; masks 1..8 stay within half)
__device__ __forceinline__ float halfmax(float v) {
    v = fmaxf(v, __shfl_xor(v, 1, 32));
    v = fmaxf(v, __shfl_xor(v, 2, 32));
    v = fmaxf(v, __shfl_xor(v, 4, 32));
    v = fmaxf(v, __shfl_xor(v, 8, 32));
    return v;
}
__device__ __forceinline__ float halfsum(float v) {
    v += __shfl_xor(v, 1, 32);
    v += __shfl_xor(v, 2, 32);
    v += __shfl_xor(v, 4, 32);
    v += __shfl_xor(v, 8, 32);
    return v;
}

#if defined(HAVE_TDM)
// ---------------------------------------------------------------------------
// TDM: DMA a 32x64 bf16 tile (row-major, stride 64 elems) from global memory
// into LDS with 16B padding after each 128B row -> 144B LDS row stride
// (matches k_lds[32][72]; 36-DWORD stride = conflict-free B-fragment reads).
// D# layout per CDNA5 ISA ch. 8 (group0: count/lds/global/type,
// group1: data_size + pad + dims/strides). Groups 2/3 zero (2D tensor).
// ---------------------------------------------------------------------------
__device__ __forceinline__ void tdm_load_k_chunk(const bf16* g, bf16* lds_dst) {
    const unsigned long long ga = (unsigned long long)(const void*)g;
    const unsigned lds_off = (unsigned)(unsigned long long)(void*)lds_dst;
    u32x4 g0;
    g0[0] = 1u;                                   // count=1 (valid), user mode
    g0[1] = lds_off;                              // lds_addr (bytes)
    g0[2] = (unsigned)ga;                         // global_addr[31:0]
    g0[3] = ((unsigned)(ga >> 32) & 0x01FFFFFFu)  // global_addr[56:32]
          | (2u << 30);                           // type = 2 ("image")
    i32x8 g1;
    g1[0] = (int)((1u << 16)                      // data_size = 1 -> 2-byte elems
          |       (1u << 20)                      // pad_enable
          |       (4u << 22)                      // pad_interval: 32 DW = 128B
          |       (3u << 25));                    // pad_amount: 4 DW = 16B
    g1[1] = (int)(64u << 16);                     // tensor_dim0 = 64 (bits 79:48)
    g1[2] = (int)(32u << 16);                     // tensor_dim1 = 32 (bits 111:80)
    g1[3] = (int)(64u << 16);                     // tile_dim0 = 64   (bits 127:112)
    g1[4] = (int)(32u);                           // tile_dim1 = 32, tile_dim2 = 0
    g1[5] = (int)(64);                            // tensor_dim0_stride = 64 elems
    g1[6] = 0;
    g1[7] = 0;
    const i32x4 z4 = {0, 0, 0, 0};
#  if defined(TDM_SIX_ARGS)
    const i32x8 z8 = {0, 0, 0, 0, 0, 0, 0, 0};
    __builtin_amdgcn_tensor_load_to_lds(g0, g1, z4, z4, z8, 0);
#  else
    __builtin_amdgcn_tensor_load_to_lds(g0, g1, z4, z4, 0);
#  endif
}
#endif

// ---------------------------------------------------------------------------
// Kernel 1: QKV projection. x(4096x1024 f32) @ Wqkv(1024x3072 f32) + bqkv.
// Output scattered to bf16 q/k/v buffers laid out (B,H,N,64); q pre-scaled
// by 1/sqrt(64) = 0.125 (exact in bf16).
// Block: 256 threads = 8 waves, tile 128(M) x 64(N), K staged in 32-chunks.
// ---------------------------------------------------------------------------
__global__ void __launch_bounds__(256)
qkv_gemm_kernel(const float* __restrict__ x, const float* __restrict__ Wqkv,
                const float* __restrict__ bqkv,
                bf16* __restrict__ qb, bf16* __restrict__ kb, bf16* __restrict__ vb)
{
    __shared__ bf16 A_lds[128][40];   // 128 rows x 32 K (bf16), padded rows (80B)
    __shared__ bf16 Bt_lds[64][40];   // transposed: [n][k], contiguous K per column

    const int tid  = threadIdx.x;
    const int lane = tid & 31, w = tid >> 5;
    const int l16  = lane & 15, hic = lane >> 4;
    const int wm   = w >> 1,    wn = w & 1;
    const int m0   = blockIdx.y * 128;
    const int n0   = blockIdx.x * 64;

    v8f acc[2][2] = {};

    for (int kk = 0; kk < DMODEL; kk += 32) {
        __syncthreads();
        // ---- stage A tile: x[m0..m0+127][kk..kk+31] -> bf16
        {
            const int c4 = (tid & 7) * 4;
            #pragma unroll
            for (int i = 0; i < 4; ++i) {
                const int r = (tid >> 3) + i * 32;
                float4 f = *(const float4*)(x + (size_t)(m0 + r) * DMODEL + kk + c4);
                v4bf h4; h4[0]=(bf16)f.x; h4[1]=(bf16)f.y; h4[2]=(bf16)f.z; h4[3]=(bf16)f.w;
                *(v4bf*)&A_lds[r][c4] = h4;
            }
        }
        // ---- stage B tile transposed: Wqkv[kk..+31][n0..+63] -> Bt[n][k]
        #pragma unroll
        for (int i = 0; i < 2; ++i) {
            const int g  = tid + i * 256;
            const int kr = g >> 4, c4 = (g & 15) * 4;
            float4 f = *(const float4*)(Wqkv + (size_t)(kk + kr) * (3*DMODEL) + n0 + c4);
            Bt_lds[c4 + 0][kr] = (bf16)f.x; Bt_lds[c4 + 1][kr] = (bf16)f.y;
            Bt_lds[c4 + 2][kr] = (bf16)f.z; Bt_lds[c4 + 3][kr] = (bf16)f.w;
        }
        __syncthreads();

        // ---- fragments + WMMA (16x16x32 bf16, f32 acc)
        const int a0 = hic ? 8 : 0;      // A: K {0..7,16..23} lo / {8..15,24..31} hi
        const int b0 = hic ? 8 : 0;      // B: K {0..7,16..23} lo / {8..15,24..31} hi
        v16bf aF[2], bF[2];
        #pragma unroll
        for (int mi = 0; mi < 2; ++mi) {
            const bf16* ar = &A_lds[wm * 32 + mi * 16 + l16][0];
            aF[mi] = bcat(ld8(ar + a0), ld8(ar + a0 + 16));
        }
        #pragma unroll
        for (int ni = 0; ni < 2; ++ni) {
            const bf16* br = &Bt_lds[wn * 32 + ni * 16 + l16][0];
            bF[ni] = bcat(ld8(br + b0), ld8(br + b0 + 16));
        }
        #pragma unroll
        for (int mi = 0; mi < 2; ++mi)
            #pragma unroll
            for (int ni = 0; ni < 2; ++ni)
                acc[mi][ni] = wmma_bf16(aF[mi], bF[ni], acc[mi][ni]);
    }

    // ---- epilogue: bias, q-scale, scatter to (B,H,N,64) bf16
    #pragma unroll
    for (int mi = 0; mi < 2; ++mi) {
        #pragma unroll
        for (int ni = 0; ni < 2; ++ni) {
            const int ncol = n0 + wn * 32 + ni * 16 + l16;
            const int sec  = ncol >> 10;           // 0=q,1=k,2=v (uniform per tile)
            const int didx = ncol & 1023;
            const int head = didx >> 6, hdc = didx & 63;
            bf16* dst = (sec == 0) ? qb : ((sec == 1) ? kb : vb);
            const float scale = (sec == 0) ? 0.125f : 1.0f;
            const float bia = bqkv[ncol];
            #pragma unroll
            for (int r = 0; r < 8; ++r) {
                const int mrow = m0 + wm * 32 + mi * 16 + r + 8 * hic;  // C: M=r+8*hi
                const int bb = mrow >> 11, nn = mrow & 2047;
                const float vv = (acc[mi][ni][r] + bia) * scale;
                dst[(((size_t)bb * NHEAD + head) * SEQ + nn) * HDIM + hdc] = (bf16)vv;
            }
        }
    }
}

// ---------------------------------------------------------------------------
// Kernel 2: coord_proj[b,h,n] = coords[b,n,:] . rel_weight[h,:]   (B*H*N vals)
// ---------------------------------------------------------------------------
__global__ void __launch_bounds__(256)
coord_proj_kernel(const float* __restrict__ coords, const float* __restrict__ rw,
                  float* __restrict__ cp)
{
    const int idx = blockIdx.x * 256 + threadIdx.x;   // 0 .. B*H*N-1 (65536)
    const int b = idx >> 15;
    const int h = (idx >> 11) & (NHEAD - 1);
    const int n = idx & (SEQ - 1);
    const float* c = coords + ((size_t)b * SEQ + n) * 3;
    const float* wp = rw + h * 3;
    cp[idx] = c[0] * wp[0] + c[1] * wp[1] + c[2] * wp[2];
}

// ---------------------------------------------------------------------------
// Kernel 3: flash attention with additive biases.
// Block = 256 threads (8 waves) per (b,h, 128 q-rows). Wave owns 16 q-rows,
// Q fragments resident in VGPRs. K chunk staged by the Tensor Data Mover
// (wave 0 issues, TENSORcnt-tracked, LDS row padding done by the TDM) while
// all threads stage V transposed; P goes C-layout -> A-layout via a per-wave
// LDS patch guarded by s_wait_dscnt.
// ---------------------------------------------------------------------------
__global__ void __launch_bounds__(256)
flash_attn_kernel(const bf16* __restrict__ qb, const bf16* __restrict__ kb,
                  const bf16* __restrict__ vb, const float* __restrict__ cp,
                  const float* __restrict__ bias, float* __restrict__ out)
{
    __shared__ bf16 k_lds[32][72];        // [k_local][hd]   row 144B (TDM-padded)
    __shared__ bf16 vT_lds[64][40];       // [hd][k_local]   row 80B
    __shared__ bf16 p_lds[8][16][40];     // per-wave P patch [M][k_local]

    const int tid  = threadIdx.x;
    const int lane = tid & 31, w = tid >> 5;
    const int l16  = lane & 15, hic = lane >> 4;
    const int bh   = blockIdx.y;                 // b*16 + h
    const int b    = bh >> 4, h = bh & 15;
    const int qbase = blockIdx.x * 128;

    // ---- preload Q fragments (A-layout: row = lane%16, K split by half-wave)
    const int qn = qbase + w * 16 + l16;
    const bf16* qrow = qb + ((size_t)bh * SEQ + qn) * HDIM;
    const int a0 = hic ? 8 : 0;
    const v16bf qA0 = bcat(ld8(qrow + a0),      ld8(qrow + a0 + 16));       // hd 0..31
    const v16bf qA1 = bcat(ld8(qrow + 32 + a0), ld8(qrow + 32 + a0 + 16));  // hd 32..63

    // ---- per-row (C-layout) state: running max / sum / coord bias of q-row
    float cq[8], mrun[8], lrun[8];
    v8f o[4] = {};
    #pragma unroll
    for (int r = 0; r < 8; ++r) {
        const int qr = qbase + w * 16 + r + 8 * hic;
        cq[r]   = cp[(size_t)bh * SEQ + qr];
        mrun[r] = -1e30f;
        lrun[r] = 0.0f;
    }

    for (int kc = 0; kc < SEQ; kc += 32) {
        __syncthreads();
#if defined(HAVE_TDM)
        // ---- K chunk via Tensor Data Mover (one DMA, wave 0 issues)
        if (w == 0)
            tdm_load_k_chunk(kb + ((size_t)bh * SEQ + kc) * HDIM, &k_lds[0][0]);
        // ---- V chunk staged transposed by all threads (overlaps the DMA)
        {
            const int r = tid >> 3, c8 = (tid & 7) * 8;
            const size_t base = ((size_t)bh * SEQ + kc + r) * HDIM + c8;
            v8bf vv = ld8(vb + base);
            #pragma unroll
            for (int j = 0; j < 8; ++j) vT_lds[c8 + j][r] = vv[j];
        }
        // ---- prefetch next chunk's bias columns into GL2 while the DMA runs
        if (kc + 32 < SEQ)
            __builtin_prefetch(bias + (size_t)qn * SEQ + kc + 32, 0, 1);
        if (w == 0)
            __builtin_amdgcn_s_wait_tensorcnt(0);
#else
        // ---- fallback: manual staging of K (raw) and V (transposed)
        {
            const int r = tid >> 3, c8 = (tid & 7) * 8;
            const size_t base = ((size_t)bh * SEQ + kc + r) * HDIM + c8;
            *(v8bf*)&k_lds[r][c8] = ld8(kb + base);
            v8bf vv = ld8(vb + base);
            #pragma unroll
            for (int j = 0; j < 8; ++j) vT_lds[c8 + j][r] = vv[j];
        }
#endif
        __syncthreads();

        // ---- scores: two 16x16 tiles (k_local 0..15, 16..31), K-dim = hd (2 steps)
        v8f s[2];
        const int kbo = hic ? 8 : 0;
        #pragma unroll
        for (int t = 0; t < 2; ++t) {
            const bf16* kr0 = &k_lds[t * 16 + l16][0];
            const v16bf kB0 = bcat(ld8(kr0 + kbo),      ld8(kr0 + kbo + 16));       // hd 0..31
            const v16bf kB1 = bcat(ld8(kr0 + 32 + kbo), ld8(kr0 + 32 + kbo + 16));  // hd 32..63
            v8f a = {};
            a = wmma_bf16(qA0, kB0, a);
            a = wmma_bf16(qA1, kB1, a);
            // additive biases: +coord_q - coord_k + attn_bias[qr][kcol]
            const int kcol = kc + t * 16 + l16;
            const float ck = cp[(size_t)bh * SEQ + kcol];
            #pragma unroll
            for (int r = 0; r < 8; ++r) {
                const int qr = qbase + w * 16 + r + 8 * hic;
                a[r] += cq[r] - ck + bias[(size_t)qr * SEQ + kcol];
            }
            s[t] = a;
        }

        // ---- online softmax (row = 16 lanes of one half-wave, per vgpr r)
        #pragma unroll
        for (int r = 0; r < 8; ++r) {
            float cmax = halfmax(fmaxf(s[0][r], s[1][r]));
            const float mn = fmaxf(mrun[r], cmax);
            const float alpha = __expf(mrun[r] - mn);
            const float p0 = __expf(s[0][r] - mn);
            const float p1 = __expf(s[1][r] - mn);
            lrun[r] = lrun[r] * alpha + halfsum(p0 + p1);
            mrun[r] = mn;
            s[0][r] = p0; s[1][r] = p1;
            o[0][r] *= alpha; o[1][r] *= alpha; o[2][r] *= alpha; o[3][r] *= alpha;
        }

        // ---- P: C-layout -> A-layout via per-wave LDS patch (bf16)
        #pragma unroll
        for (int r = 0; r < 8; ++r) {
            p_lds[w][r + 8 * hic][l16]      = (bf16)s[0][r];
            p_lds[w][r + 8 * hic][16 + l16] = (bf16)s[1][r];
        }
        asm volatile("s_wait_dscnt 0" ::: "memory");   // wave-local LDS RAW fence
        const int pk = hic ? 8 : 0;
        const bf16* pr = &p_lds[w][l16][0];
        const v16bf pA = bcat(ld8(pr + pk), ld8(pr + pk + 16));

        // ---- PV: 4 output tiles over hd, B from transposed V (K {0..7,16..23} lo / {8..15,24..31} hi)
        const int vkb = hic ? 8 : 0;
        #pragma unroll
        for (int hdt = 0; hdt < 4; ++hdt) {
            const bf16* vr = &vT_lds[hdt * 16 + l16][0];
            const v16bf vB = bcat(ld8(vr + vkb), ld8(vr + vkb + 16));
            o[hdt] = wmma_bf16(pA, vB, o[hdt]);
        }
    }

    // ---- epilogue: normalize, write out (B,N,H,hd) f32
    #pragma unroll
    for (int r = 0; r < 8; ++r) {
        const float rl = 1.0f / lrun[r];
        const int qr = qbase + w * 16 + r + 8 * hic;
        float* orow = out + ((size_t)b * SEQ + qr) * DMODEL + h * HDIM;
        #pragma unroll
        for (int hdt = 0; hdt < 4; ++hdt)
            orow[hdt * 16 + l16] = o[hdt][r] * rl;
    }
}

// ---------------------------------------------------------------------------
// Host launcher
// ---------------------------------------------------------------------------
extern "C" void kernel_launch(void* const* d_in, const int* in_sizes, int n_in,
                              void* d_out, int out_size, void* d_ws, size_t ws_size,
                              hipStream_t stream) {
    (void)in_sizes; (void)n_in; (void)out_size; (void)ws_size;
    const float* x         = (const float*)d_in[0];
    const float* coords    = (const float*)d_in[1];
    const float* attn_bias = (const float*)d_in[2];
    const float* Wqkv      = (const float*)d_in[3];
    const float* bqkv      = (const float*)d_in[4];
    const float* relw      = (const float*)d_in[5];
    float* out = (float*)d_out;

    // workspace: q/k/v bf16 (B,H,N,64) = 4 Mi elems each, + coord_proj f32
    const size_t QSZ = (size_t)BATCH * NHEAD * SEQ * HDIM;   // 4,194,304
    bf16* qb = (bf16*)d_ws;
    bf16* kb = qb + QSZ;
    bf16* vb = kb + QSZ;
    float* cp = (float*)(vb + QSZ);

    qkv_gemm_kernel<<<dim3((3 * DMODEL) / 64, (BATCH * SEQ) / 128), 256, 0, stream>>>(
        x, Wqkv, bqkv, qb, kb, vb);
    coord_proj_kernel<<<dim3((BATCH * NHEAD * SEQ) / 256), 256, 0, stream>>>(
        coords, relw, cp);
    flash_attn_kernel<<<dim3(SEQ / 128, BATCH * NHEAD), 256, 0, stream>>>(
        qb, kb, vb, cp, attn_bias, out);
}
